// TransducerBeamSearcher_24713241821299
// MI455X (gfx1250) — matrix operand order repaired and checked by reading, not compiled, not run on hardware
//
#include <hip/hip_runtime.h>

typedef _Float16 half_t;
typedef __attribute__((ext_vector_type(16))) _Float16 v16h;
typedef __attribute__((ext_vector_type(8)))  _Float16 v8h;
typedef __attribute__((ext_vector_type(8)))  float    v8f;

#define BB 8
#define TT 128
#define HH 512
#define VV 4096
#define EE 512
#define JJ 512
#define KB 8
#define NBEST 4
#define UMAXC 64
#define NEGC (-1.0e4f)
#define FNEG (-3.0e38f)

// ---------------- LDS layout (bytes) ----------------
#define L_X      0                  // half x[16][512]   (A for gi GEMM)
#define L_H      16384              // half h[16][512]   (A for gh GEMM, then hnew_f16 for joint)
#define L_J      32768              // half j[16][512]   (A for output GEMM)
#define L_HID    49152              // float hid[2][8][512] (double-buffered hidden)
#define L_HNEW   81920              // float hnew[8][512]
#define L_Q      98304              // union: gi[8][1536]+gh[8][1536] | jt[8][512] | logits[8][4096]
#define L_REDV   229376             // float red[1024]
#define L_REDI   233472             // int   red[1024]
#define L_PREDS  237568             // int preds[2][8][64]
#define L_SMALL  241664             // small state
#define LDS_BYTES 242176

// ------------- fp32 -> f16 elementwise -------------
__global__ void cvt_f16_kernel(const float* __restrict__ src, half_t* __restrict__ dst, int n) {
  int i = blockIdx.x * blockDim.x + threadIdx.x;
  if (i < n) dst[i] = (half_t)src[i];
}

// ------------- fp32 weight -> f16 WMMA-B-fragment swizzle -------------
// Output layout: tile-major, tile = nt*(K/32)+kt; within tile 32 lanes x 16 halfs
// contiguous. Lane l supplies column n = nt*16 + (l&15); element e -> K row
// kt*32 + ((l&16)?8:0) + (e&7) + ((e&8)?16:0)  (lanes 0-15 hold K {0..7,16..23}, 16-31 hold K {8..15,24..31}).
__global__ void swizzle_b_kernel(const float* __restrict__ w, half_t* __restrict__ dst,
                                 int K, int N) {
  int idx = blockIdx.x * blockDim.x + threadIdx.x;
  if (idx >= K * N) return;
  int e    = idx & 15;
  int lane = (idx >> 4) & 31;
  int tile = idx >> 9;
  int KT = K >> 5;
  int kt = tile % KT;
  int nt = tile / KT;
  int n = (nt << 4) + (lane & 15);
  int k = (kt << 5) + ((lane & 16) ? 8 : 0) + (e & 7) + ((e & 8) ? 16 : 0);
  dst[idx] = (half_t)w[(size_t)k * N + n];
}

// ------------- WMMA helpers -------------
// A fragment (16-bit A 16x32, ISA 7.12.2): lane l -> row l&15; lanes 0-15 hold
// K {0..7,16..23}, lanes 16-31 hold K {8..15,24..31}. Two contiguous 16B loads.
__device__ __forceinline__ v16h load_a_frag(const half_t* __restrict__ A, int lane, int k0) {
  int row  = lane & 15;
  int base = (lane & 16) ? 8 : 0;
  const half_t* p = A + row * 512 + k0 + base;
  v8h lo = *(const v8h*)p;
  v8h hi = *(const v8h*)(p + 16);
  v16h a;
#pragma unroll
  for (int i = 0; i < 8; ++i) { a[i] = lo[i]; a[i + 8] = hi[i]; }
  return a;
}

__device__ __forceinline__ v16h load_b_frag(const half_t* __restrict__ Wsw, int tile, int lane) {
  return *(const v16h*)(Wsw + (((size_t)tile * 32 + lane) << 4));
}

__device__ __forceinline__ v8f wmma_f16(v16h a, v16h b, v8f c) {
  return __builtin_amdgcn_wmma_f32_16x16x32_f16(false, a, false, b, (short)0, c, false, false);
}

// ------------- tn_proj[b,t,:] = tn[b,t,:] @ w_tn  (hoisted out of time loop) -------------
__global__ void __launch_bounds__(32) tn_proj_kernel(const half_t* __restrict__ tnf,
                                                     const half_t* __restrict__ wtn_sw,
                                                     float* __restrict__ out) {
  int lane = threadIdx.x;
  int m0 = blockIdx.x << 4;   // rows of [B*T]
  int n0 = blockIdx.y << 4;
  const half_t* A = tnf + (size_t)m0 * HH;
  v8f acc = {};
  for (int kt = 0; kt < HH / 32; ++kt) {
    v16h a = load_a_frag(A, lane, kt << 5);
    v16h b = load_b_frag(wtn_sw, blockIdx.y * (HH / 32) + kt, lane);
    acc = wmma_f16(a, b, acc);
  }
  int col = lane & 15;
  int rbase = (lane & 16) ? 8 : 0;   // C/D: lanes 0-15 -> M 0-7, lanes 16-31 -> M 8-15
#pragma unroll
  for (int r = 0; r < 8; ++r)
    out[(size_t)(m0 + rbase + r) * JJ + n0 + col] = acc[r];
}

// ------------- persistent beam-search decode: 1 block per batch -------------
__global__ void __launch_bounds__(1024) rnnt_decode_kernel(
    const half_t* __restrict__ embf,     // [V][E] f16
    const half_t* __restrict__ wih_sw,   // swizzled [512 x 1536]
    const half_t* __restrict__ whh_sw,   // swizzled [512 x 1536]
    const half_t* __restrict__ wpn_sw,   // swizzled [512 x 512]
    const half_t* __restrict__ wout_sw,  // swizzled [512 x 4096]
    const float*  __restrict__ tnproj,   // [B][T][J] f32
    const float*  __restrict__ b_gru,    // [1536]
    const float*  __restrict__ b_joint,  // [512]
    const float*  __restrict__ b_out,    // [4096]
    float* __restrict__ outbuf)          // preds B*NBEST*UMAX, then scores B*NBEST
{
  extern __shared__ __align__(128) unsigned char sm[];
  half_t* xA  = (half_t*)(sm + L_X);
  half_t* hA  = (half_t*)(sm + L_H);
  half_t* jA  = (half_t*)(sm + L_J);
  float* hid  = (float*)(sm + L_HID);
  float* hnw  = (float*)(sm + L_HNEW);
  float* gi   = (float*)(sm + L_Q);
  float* gh   = gi + 8 * 1536;
  float* jt   = (float*)(sm + L_Q);
  float* lg   = (float*)(sm + L_Q);
  float* redv = (float*)(sm + L_REDV);
  int*   redi = (int*)(sm + L_REDI);
  int*   prd  = (int*)(sm + L_PREDS);
  int*   tok   = (int*)(sm + L_SMALL);        // [2][8]
  int*   lenb  = tok + 16;                    // [2][8]
  float* scl   = (float*)(lenb + 16);         // [8]
  float* selv  = scl + 8;                     // [8]
  int*   seli  = (int*)(selv + 8);            // [8]
  float* radj  = (float*)(seli + 8);          // [8]
  int*   order = (int*)(radj + 8);            // [8]
  float* normv = (float*)(order + 8);         // [8]

  const int tid  = threadIdx.x;
  const int lane = tid & 31;
  const int wid  = tid >> 5;
  const int b    = blockIdx.x;

  // ---- init beam state (buffer 0), zero pad rows 8..15 of A buffers ----
  for (int i = tid; i < 8 * 512; i += 1024) hid[i] = 0.0f;
  for (int i = tid; i < 8 * 512; i += 1024) {
    int r = 8 + (i >> 9), c = i & 511;
    xA[r * 512 + c] = (half_t)0.0f;
    hA[r * 512 + c] = (half_t)0.0f;
    jA[r * 512 + c] = (half_t)0.0f;
  }
  for (int i = tid; i < 8 * 64; i += 1024) prd[i] = 0;
  if (tid < 8) {
    tok[tid]  = 0;                       // BOS
    lenb[tid] = 1;
    scl[tid]  = (tid == 0) ? 0.0f : NEGC;
  }
  __syncthreads();

  for (int t = 0; t < TT; ++t) {
    const int cur = t & 1, nxt = cur ^ 1;
    int*   tok_c = tok + cur * 8;   int*   tok_n = tok + nxt * 8;
    int*   len_c = lenb + cur * 8;  int*   len_n = lenb + nxt * 8;
    float* hid_c = hid + cur * 8 * 512;
    float* hid_n = hid + nxt * 8 * 512;
    int*   prd_c = prd + cur * 8 * 64;
    int*   prd_n = prd + nxt * 8 * 64;

    // phase 0: stage x = emb[tokens] and h (f32->f16) into LDS A buffers
    for (int i = tid; i < 8 * 512; i += 1024) {
      int k = i >> 9, c = i & 511;
      xA[k * 512 + c] = embf[(size_t)tok_c[k] * EE + c];
      hA[k * 512 + c] = (half_t)hid_c[i];
    }
    __syncthreads();

    // phase 1: gi = x @ w_ih, gh = h @ w_hh   (192 16x16 tiles, 6 per wave)
    for (int tix = 0; tix < 6; ++tix) {
      int id = wid * 6 + tix;
      const half_t* Ab; const half_t* Wb; float* Ob; int nt;
      if (id < 96) { Ab = xA; Wb = wih_sw; Ob = gi; nt = id; }
      else         { Ab = hA; Wb = whh_sw; Ob = gh; nt = id - 96; }
      v8f acc = {};
      for (int kt = 0; kt < 16; ++kt) {
        v16h a = load_a_frag(Ab, lane, kt << 5);
        v16h w = load_b_frag(Wb, nt * 16 + kt, lane);
        acc = wmma_f16(a, w, acc);
      }
      if (lane < 16) {
#pragma unroll
        for (int r = 0; r < 8; ++r) Ob[r * 1536 + nt * 16 + lane] = acc[r];
      }
    }
    __syncthreads();

    // phase 2: GRU gates -> hnew (f32) and hnew_f16 (into hA for joint GEMM)
    for (int i = tid; i < 8 * 512; i += 1024) {
      int k = i >> 9, c = i & 511;
      float ir = gi[k * 1536 + c]          + b_gru[c];
      float iz = gi[k * 1536 + c + 512]    + b_gru[c + 512];
      float in_ = gi[k * 1536 + c + 1024]  + b_gru[c + 1024];
      float hr = gh[k * 1536 + c];
      float hz = gh[k * 1536 + c + 512];
      float hn = gh[k * 1536 + c + 1024];
      float r = 1.0f / (1.0f + expf(-(ir + hr)));
      float z = 1.0f / (1.0f + expf(-(iz + hz)));
      float n = tanhf(in_ + r * hn);
      float hv = (1.0f - z) * n + z * hid_c[i];
      hnw[i] = hv;
      hA[k * 512 + c] = (half_t)hv;
    }
    __syncthreads();

    // phase 3: jt = hnew @ w_pn   (32 tiles, 1 per wave)
    {
      int nt = wid;
      v8f acc = {};
      for (int kt = 0; kt < 16; ++kt) {
        v16h a = load_a_frag(hA, lane, kt << 5);
        v16h w = load_b_frag(wpn_sw, nt * 16 + kt, lane);
        acc = wmma_f16(a, w, acc);
      }
      if (lane < 16) {
#pragma unroll
        for (int r = 0; r < 8; ++r) jt[r * 512 + nt * 16 + lane] = acc[r];
      }
    }
    __syncthreads();

    // phase 4: j = tanh(tn_proj + pn_proj + b_joint) -> f16
    for (int i = tid; i < 8 * 512; i += 1024) {
      int k = i >> 9, c = i & 511;
      float v = tanhf(tnproj[((size_t)b * TT + t) * JJ + c] + jt[i] + b_joint[c]);
      jA[k * 512 + c] = (half_t)v;
    }
    __syncthreads();

    // phase 5: logits = j @ w_out + b_out   (256 tiles, 8 per wave)
    for (int tix = 0; tix < 8; ++tix) {
      int nt = wid * 8 + tix;
      v8f acc = {};
      for (int kt = 0; kt < 16; ++kt) {
        v16h a = load_a_frag(jA, lane, kt << 5);
        v16h w = load_b_frag(wout_sw, nt * 16 + kt, lane);
        acc = wmma_f16(a, w, acc);
      }
      if (lane < 16) {
        int cc = nt * 16 + lane;
        float bo = b_out[cc];
#pragma unroll
        for (int r = 0; r < 8; ++r) lg[r * VV + cc] = acc[r] + bo;
      }
    }
    __syncthreads();

    // phase 6: per-row log-softmax fold + beam score:  lg[k,v] += scores[k]-M_k-logSumExp_k
    {
      int k = tid >> 7, g = tid & 127;
      float m = FNEG;
      for (int c = g; c < VV; c += 128) m = fmaxf(m, lg[k * VV + c]);
      redv[tid] = m; __syncthreads();
      for (int s = 64; s > 0; s >>= 1) {
        if (g < s) redv[tid] = fmaxf(redv[tid], redv[tid + s]);
        __syncthreads();
      }
      float rowm = redv[k << 7];
      __syncthreads();
      float sum = 0.0f;
      for (int c = g; c < VV; c += 128) sum += expf(lg[k * VV + c] - rowm);
      redv[tid] = sum; __syncthreads();
      for (int s = 64; s > 0; s >>= 1) {
        if (g < s) redv[tid] += redv[tid + s];
        __syncthreads();
      }
      if (g == 0) radj[k] = scl[k] - rowm - logf(redv[k << 7]);
      __syncthreads();
      for (int i = tid; i < KB * VV; i += 1024) lg[i] += radj[i >> 12];
    }
    __syncthreads();

    // phase 7: top-8 over K*V (iterative argmax; ties -> lowest flat index, jax-compatible)
    for (int it = 0; it < KB; ++it) {
      float bv = FNEG; int bi = 0;
      for (int i = tid; i < KB * VV; i += 1024) {
        float v = lg[i];
        if (v > bv) { bv = v; bi = i; }
      }
      redv[tid] = bv; redi[tid] = bi;
      __syncthreads();
      for (int s = 512; s > 0; s >>= 1) {
        if (tid < s) {
          float ov = redv[tid + s]; int oi = redi[tid + s];
          if (ov > redv[tid] || (ov == redv[tid] && oi < redi[tid])) {
            redv[tid] = ov; redi[tid] = oi;
          }
        }
        __syncthreads();
      }
      if (tid == 0) {
        selv[it] = redv[0]; seli[it] = redi[0];
        lg[redi[0]] = FNEG;
      }
      __syncthreads();
    }

    // phase 8: beam update (gather parent state; blank keeps parent hidden/pred)
    for (int i = tid; i < KB * 512; i += 1024) {
      int k = i >> 9, c = i & 511;
      int si = seli[k]; int parent = si >> 12; int tk = si & (VV - 1);
      hid_n[k * 512 + c] = (tk == 0) ? hid_c[parent * 512 + c] : hnw[parent * 512 + c];
    }
    for (int i = tid; i < KB * UMAXC; i += 1024) {
      int k = i / UMAXC, u = i % UMAXC;
      int parent = seli[k] >> 12;
      prd_n[k * UMAXC + u] = prd_c[parent * UMAXC + u];
    }
    __syncthreads();
    if (tid < KB) {
      int si = seli[tid]; int parent = si >> 12; int tk = si & (VV - 1);
      int pl = len_c[parent];
      bool blank = (tk == 0);
      tok_n[tid] = blank ? tok_c[parent] : tk;
      len_n[tid] = pl + (blank ? 0 : 1);
      scl[tid]   = selv[tid];
      if (!blank) {
        int wp = pl - 1;
        if (wp >= 0 && wp < UMAXC) prd_n[tid * UMAXC + wp] = tk;
      }
    }
    __syncthreads();
  }

  // ---- finalize: norm = score/len, stable top-NBEST, write outputs ----
  const int fin = TT & 1;   // == 0
  if (tid < KB) normv[tid] = scl[tid] / (float)lenb[fin * 8 + tid];
  __syncthreads();
  if (tid == 0) {
    unsigned used = 0;
    for (int j = 0; j < NBEST; ++j) {
      int bi = -1; float bv = FNEG;
      for (int k = 0; k < KB; ++k) {
        if (used & (1u << k)) continue;
        if (bi < 0 || normv[k] > bv) { bv = normv[k]; bi = k; }
      }
      used |= 1u << bi;
      order[j] = bi;
    }
  }
  __syncthreads();
  float* outp = outbuf;                               // [B][NBEST][UMAX]
  float* outs = outbuf + (size_t)BB * NBEST * UMAXC;  // [B][NBEST]
  for (int i = tid; i < NBEST * UMAXC; i += 1024) {
    int j = i / UMAXC, u = i % UMAXC;
    outp[((size_t)b * NBEST + j) * UMAXC + u] =
        (float)prd[fin * 8 * 64 + order[j] * UMAXC + u];
  }
  if (tid < NBEST) outs[b * NBEST + tid] = normv[order[tid]];
}

extern "C" void kernel_launch(void* const* d_in, const int* in_sizes, int n_in,
                              void* d_out, int out_size, void* d_ws, size_t ws_size,
                              hipStream_t stream) {
  (void)in_sizes; (void)n_in; (void)out_size; (void)ws_size;
  const float* tn   = (const float*)d_in[0];   // [B,T,H]
  const float* emb  = (const float*)d_in[1];   // [V,E]
  const float* wih  = (const float*)d_in[2];   // [E,3Hd]
  const float* whh  = (const float*)d_in[3];   // [Hd,3Hd]
  const float* bgru = (const float*)d_in[4];   // [3Hd]
  const float* wtn  = (const float*)d_in[5];   // [H,J]
  const float* wpn  = (const float*)d_in[6];   // [Hd,J]
  const float* bj   = (const float*)d_in[7];   // [J]
  const float* wout = (const float*)d_in[8];   // [J,V]
  const float* bout = (const float*)d_in[9];   // [V]
  float* out = (float*)d_out;

  // workspace carve (all region sizes are multiples of 32B)
  half_t* emb_f  = (half_t*)d_ws;                          // 4096*512
  half_t* tn_f   = emb_f  + (size_t)VV * EE;               // 8*128*512
  half_t* wih_s  = tn_f   + (size_t)BB * TT * HH;          // 512*1536
  half_t* whh_s  = wih_s  + (size_t)512 * 1536;
  half_t* wpn_s  = whh_s  + (size_t)512 * 1536;            // 512*512
  half_t* wtn_s  = wpn_s  + (size_t)512 * 512;             // 512*512
  half_t* wout_s = wtn_s  + (size_t)512 * 512;             // 512*4096
  float*  tnp    = (float*)(wout_s + (size_t)512 * VV);    // 8*128*512 f32

  int n;
  n = VV * EE;          cvt_f16_kernel<<<(n + 255) / 256, 256, 0, stream>>>(emb, emb_f, n);
  n = BB * TT * HH;     cvt_f16_kernel<<<(n + 255) / 256, 256, 0, stream>>>(tn, tn_f, n);
  n = 512 * 1536;       swizzle_b_kernel<<<(n + 255) / 256, 256, 0, stream>>>(wih, wih_s, 512, 1536);
                        swizzle_b_kernel<<<(n + 255) / 256, 256, 0, stream>>>(whh, whh_s, 512, 1536);
  n = 512 * 512;        swizzle_b_kernel<<<(n + 255) / 256, 256, 0, stream>>>(wpn, wpn_s, 512, 512);
                        swizzle_b_kernel<<<(n + 255) / 256, 256, 0, stream>>>(wtn, wtn_s, 512, 512);
  n = 512 * VV;         swizzle_b_kernel<<<(n + 255) / 256, 256, 0, stream>>>(wout, wout_s, 512, VV);

  tn_proj_kernel<<<dim3((BB * TT) / 16, JJ / 16), 32, 0, stream>>>(tn_f, wtn_s, tnp);

  rnnt_decode_kernel<<<dim3(BB), dim3(1024), LDS_BYTES, stream>>>(
      emb_f, wih_s, whh_s, wpn_s, wout_s, tnp, bgru, bj, bout, out);
}
